// Encoder_20976620274135
// MI455X (gfx1250) — matrix operand order repaired and checked by reading, not compiled, not run on hardware
//
#include <hip/hip_runtime.h>
#include <math.h>

typedef __attribute__((ext_vector_type(16))) _Float16 v16h;
typedef __attribute__((ext_vector_type(8)))  float    v8f;

#define WMMA(a, b, c) __builtin_amdgcn_wmma_f32_16x16x32_f16(false, (a), false, (b), (short)0, (c), false, false)

static __device__ __forceinline__ float half16_reduce_max(float v) {
  #pragma unroll
  for (int m = 8; m >= 1; m >>= 1) v = fmaxf(v, __shfl_xor(v, m, 32));
  return v;
}
static __device__ __forceinline__ float half16_reduce_sum(float v) {
  #pragma unroll
  for (int m = 8; m >= 1; m >>= 1) v += __shfl_xor(v, m, 32);
  return v;
}
static __device__ __forceinline__ float gelu_tanh(float x) {
  float x3 = x * x * x;
  return 0.5f * x * (1.0f + tanhf(0.7978845608028654f * (x + 0.044715f * x3)));
}

// -------------------- LayerNorm (affine-free, biased var) --------------------
__global__ __launch_bounds__(256)
void ln_kernel(const float* __restrict__ in, float* __restrict__ out, int S, int D) {
  int row  = blockIdx.x * 8 + (threadIdx.x >> 5);
  int lane = threadIdx.x & 31;
  if (row >= S) return;
  const float* r = in + (size_t)row * D;            // D == 256
  float xs[8], sum = 0.f, sq = 0.f;
  #pragma unroll
  for (int i = 0; i < 8; ++i) {
    float xv = r[lane + 32 * i];
    xs[i] = xv; sum += xv; sq += xv * xv;
  }
  #pragma unroll
  for (int m = 16; m >= 1; m >>= 1) { sum += __shfl_xor(sum, m, 32); sq += __shfl_xor(sq, m, 32); }
  float mean = sum / (float)D;
  float var  = sq / (float)D - mean * mean;
  float inv  = rsqrtf(var + 1e-5f);
  float* w = out + (size_t)row * D;
  #pragma unroll
  for (int i = 0; i < 8; ++i) w[lane + 32 * i] = (xs[i] - mean) * inv;
}

// ------------- operand pre-pack into WMMA layouts (split hi/lo f16) ----------
// A-operand: element(m,k) = src[b*bstride + m*srow + k*scol] * scale
// stored at dst[((b*ntotal + ((mt*KC+kc)*32+lane)) * 16 + h]
// with m = mt*16+(lane&15), k = kc*32 + ((h<8)?h:h+8) + 8*(lane>=16)
__global__ __launch_bounds__(256)
void pack_a_kernel(const float* __restrict__ src, _Float16* __restrict__ dhi,
                   _Float16* __restrict__ dlo, int M, int K,
                   long srow, long scol, long bstride, float scale) {
  int KC = K >> 5;
  int total = (M >> 4) * KC * 32;
  int t = blockIdx.x * 256 + threadIdx.x;
  if (t >= total) return;
  int b = blockIdx.y;
  int lane = t & 31, lm = lane & 15, hs = lane >> 4;
  int kc = (t >> 5) % KC;
  int mt = (t >> 5) / KC;
  const float* s = src + (long)b * bstride + (long)(mt * 16 + lm) * srow;
  size_t base = ((size_t)b * total + t) * 16;
  #pragma unroll
  for (int h = 0; h < 16; ++h) {
    int k = kc * 32 + ((h < 8) ? h : h + 8) + (hs ? 8 : 0);
    float v = s[(long)k * scol] * scale;
    _Float16 hi = (_Float16)v;
    dhi[base + h] = hi;
    dlo[base + h] = (_Float16)(v - (float)hi);
  }
}
// B-operand: element(k,n) = src[b*bstride + k*srow + n*scol]
// with n = nt*16+(lane&15), k = kc*32 + ((h<8)?h:h+8) + 8*(lane>=16)
__global__ __launch_bounds__(256)
void pack_b_kernel(const float* __restrict__ src, _Float16* __restrict__ dhi,
                   _Float16* __restrict__ dlo, int K, int N,
                   long srow, long scol, long bstride) {
  int KC = K >> 5;
  int total = (N >> 4) * KC * 32;
  int t = blockIdx.x * 256 + threadIdx.x;
  if (t >= total) return;
  int b = blockIdx.y;
  int lane = t & 31, lm = lane & 15, hs = lane >> 4;
  int kc = (t >> 5) % KC;
  int nt = (t >> 5) / KC;
  const float* s = src + (long)b * bstride + (long)(nt * 16 + lm) * scol;
  size_t base = ((size_t)b * total + t) * 16;
  #pragma unroll
  for (int h = 0; h < 16; ++h) {
    int k = kc * 32 + ((h < 8) ? h : h + 8) + (hs ? 8 : 0);
    float v = s[(long)k * srow];
    _Float16 hi = (_Float16)v;
    dhi[base + h] = hi;
    dlo[base + h] = (_Float16)(v - (float)hi);
  }
}

// ---- GEMM on pre-packed operands: C = [residual +] act(A@B + bias) ----------
// inner loop: 4x contiguous v16h loads (b128 pairs) + 3 WMMAs, fp32 acc.
__global__ __launch_bounds__(256)
void gemm_packed_kernel(const _Float16* __restrict__ Ahi, const _Float16* __restrict__ Alo,
                        const _Float16* __restrict__ Bhi, const _Float16* __restrict__ Blo,
                        const float* __restrict__ bias, const float* __restrict__ residual,
                        float* __restrict__ C, int M, int N, int K, int fuse_gelu) {
  int w    = (int)((blockIdx.x * blockDim.x + threadIdx.x) >> 5);
  int lane = threadIdx.x & 31;
  int tiles_n = N >> 4;
  int tm = w / tiles_n, tn = w % tiles_n;
  int m0 = tm << 4, n0 = tn << 4;
  if (m0 >= M) return;                       // wave-uniform
  int hsel = lane >> 4;
  int lm   = lane & 15;
  int KC   = K >> 5;

  const v16h* Ah = (const v16h*)Ahi + (size_t)tm * KC * 32 + lane;
  const v16h* Al = (const v16h*)Alo + (size_t)tm * KC * 32 + lane;
  const v16h* Bh = (const v16h*)Bhi + (size_t)tn * KC * 32 + lane;
  const v16h* Bl = (const v16h*)Blo + (size_t)tn * KC * 32 + lane;

  v8f c = {};
  for (int kc = 0; kc < KC; ++kc) {
    v16h ah = Ah[kc * 32];
    v16h al = Al[kc * 32];
    v16h bh = Bh[kc * 32];
    v16h bl = Bl[kc * 32];
    c = WMMA(ah, bh, c);
    c = WMMA(al, bh, c);
    c = WMMA(ah, bl, c);
  }
  int n = n0 + lm;
  float bs = bias[n];
  #pragma unroll
  for (int r = 0; r < 8; ++r) {
    int m = m0 + r + (hsel ? 8 : 0);
    float v = c[r] + bs;
    if (fuse_gelu) v = gelu_tanh(v);
    if (residual)  v += residual[(size_t)m * N + n];
    C[(size_t)m * N + n] = v;
  }
}

// -------- sliding-window flash attention on pre-packed Q/K/V -----------------
__global__ __launch_bounds__(256)
void attn_kernel(const _Float16* __restrict__ Qhi, const _Float16* __restrict__ Qlo,
                 const _Float16* __restrict__ Khi, const _Float16* __restrict__ Klo,
                 const _Float16* __restrict__ Vhi, const _Float16* __restrict__ Vlo,
                 float* __restrict__ obuf, int S, int H, int DH, int halfwin) {
  const int D = H * DH;                      // 256
  int wid  = threadIdx.x >> 5;
  int gw   = blockIdx.x * 8 + wid;
  int qb   = gw / H;
  int head = gw % H;
  int q0   = qb << 4;
  if (q0 >= S) return;                       // wave-uniform
  int lane = threadIdx.x & 31;
  int hsel = lane >> 4;
  int lm   = lane & 15;

  __shared__ float pbuf[8][16][33];          // per-wave P transpose buffer
  float (*pb)[33] = pbuf[wid];

  // per-head packed-operand extents (in v16h units)
  const size_t qtot = (size_t)(S >> 4) * 32;         // Q A-pack, KC=1
  const size_t ktot = (size_t)(S >> 4) * 32;         // K^T B-pack, KC=1, ntiles=S/16
  const size_t vtot = (size_t)2 * (S >> 5) * 32;     // V B-pack, ntiles=2, KC=S/32

  size_t qidx = (size_t)head * qtot + (size_t)(q0 >> 4) * 32 + lane;
  v16h q_hi = ((const v16h*)Qhi)[qidx];
  v16h q_lo = ((const v16h*)Qlo)[qidx];

  v8f o0 = {}, o1 = {};
  float row_max[8], row_sum[8];
  #pragma unroll
  for (int r = 0; r < 8; ++r) { row_max[r] = -1e30f; row_sum[r] = 0.f; }

  int wstart = q0 - halfwin; if (wstart < 0) wstart = 0; wstart &= ~31;
  int wend   = q0 + 16 + halfwin; if (wend > S) wend = S;

  for (int c0 = wstart; c0 < wend; c0 += 32) {
    float s[2][8];
    #pragma unroll
    for (int ct = 0; ct < 2; ++ct) {
      size_t kidx = (size_t)head * ktot + (size_t)((c0 >> 4) + ct) * 32 + lane;
      v16h b_hi = ((const v16h*)Khi)[kidx];
      v16h b_lo = ((const v16h*)Klo)[kidx];
      v8f sc = {};
      sc = WMMA(q_hi, b_hi, sc);
      sc = WMMA(q_lo, b_hi, sc);
      sc = WMMA(q_hi, b_lo, sc);
      int ncol = c0 + ct * 16 + lm;
      #pragma unroll
      for (int r = 0; r < 8; ++r) {
        int mrow = q0 + r + (hsel ? 8 : 0);
        int dlt  = mrow - ncol;
        bool ok  = (dlt <= halfwin) && (dlt >= -halfwin) && (ncol < S);
        s[ct][r] = ok ? sc[r] : -1e30f;
      }
    }
    // online softmax update (row stats live per 16-lane half)
    float fac[8];
    #pragma unroll
    for (int r = 0; r < 8; ++r) {
      float t  = half16_reduce_max(fmaxf(s[0][r], s[1][r]));
      float mn = fmaxf(row_max[r], t);
      fac[r]   = __expf(row_max[r] - mn);
      row_max[r] = mn;
      float p0 = (s[0][r] <= -1e29f) ? 0.f : __expf(s[0][r] - mn);
      float p1 = (s[1][r] <= -1e29f) ? 0.f : __expf(s[1][r] - mn);
      int mrow = r + (hsel ? 8 : 0);
      pb[mrow][lm]      = p0;
      pb[mrow][16 + lm] = p1;
      row_sum[r] = row_sum[r] * fac[r] + half16_reduce_sum(p0 + p1);
    }
    #pragma unroll
    for (int r = 0; r < 8; ++r) { o0[r] *= fac[r]; o1[r] *= fac[r]; }

    // same-wave cross-lane LDS RAW: LDS is per-wave in-order; force the wait
    asm volatile("s_wait_dscnt 0" ::: "memory");

    // P in A-operand layout
    v16h p_hi, p_lo;
    #pragma unroll
    for (int h = 0; h < 16; ++h) {
      int kk = ((h < 8) ? h : h + 8) + (hsel ? 8 : 0);
      float pv = pb[lm][kk];
      _Float16 ph = (_Float16)pv;
      p_hi[h] = ph; p_lo[h] = (_Float16)(pv - (float)ph);
    }
    // O(16x32) += P(16x32) @ V(32x32), two 16-col d halves
    #pragma unroll
    for (int t = 0; t < 2; ++t) {
      size_t vidx = (size_t)head * vtot + ((size_t)t * (S >> 5) + (c0 >> 5)) * 32 + lane;
      v16h v_hi = ((const v16h*)Vhi)[vidx];
      v16h v_lo = ((const v16h*)Vlo)[vidx];
      v8f& oa = t ? o1 : o0;
      oa = WMMA(p_hi, v_hi, oa);
      oa = WMMA(p_lo, v_hi, oa);
      oa = WMMA(p_hi, v_lo, oa);
    }
  }

  #pragma unroll
  for (int r = 0; r < 8; ++r) {
    float inv = 1.0f / row_sum[r];
    int mrow  = q0 + r + (hsel ? 8 : 0);
    float* orow = obuf + (size_t)mrow * D + head * DH;
    orow[lm]      = o0[r] * inv;
    orow[16 + lm] = o1[r] * inv;
  }
}

// ------------------------------- host driver --------------------------------
extern "C" void kernel_launch(void* const* d_in, const int* in_sizes, int n_in,
                              void* d_out, int out_size, void* d_ws, size_t ws_size,
                              hipStream_t stream) {
  const int S = 4096, D = 256, FF = 1024, NLAYER = 4, H = 8, DH = 32, HALFWIN = 256;
  const float* x  = (const float*)d_in[0];
  const float* wq = (const float*)d_in[1];  const float* bq = (const float*)d_in[2];
  const float* wk = (const float*)d_in[3];  const float* bk = (const float*)d_in[4];
  const float* wv = (const float*)d_in[5];  const float* bv = (const float*)d_in[6];
  const float* wo = (const float*)d_in[7];  const float* bo = (const float*)d_in[8];
  const float* w1 = (const float*)d_in[9];  const float* b1 = (const float*)d_in[10];
  const float* w2 = (const float*)d_in[11]; const float* b2 = (const float*)d_in[12];

  char* p = (char*)d_ws;
  auto allocf = [&](size_t n) { float* r = (float*)p; p += n * sizeof(float); return r; };
  auto alloch = [&](size_t n) { _Float16* r = (_Float16*)p; p += n * sizeof(_Float16); return r; };

  float* h  = allocf((size_t)S * D);
  float* nb = allocf((size_t)S * D);
  float* qb = allocf((size_t)S * D);
  float* kb = allocf((size_t)S * D);
  float* vb = allocf((size_t)S * D);
  float* ob = allocf((size_t)S * D);
  float* ff = allocf((size_t)S * FF);
  _Float16* aAh = alloch((size_t)S * FF);   // shared A-pack buffer (max extent S*FF)
  _Float16* aAl = alloch((size_t)S * FF);
  _Float16* wBh = alloch((size_t)FF * D);   // shared B-pack buffer (max K*N)
  _Float16* wBl = alloch((size_t)FF * D);
  _Float16* qPh = alloch((size_t)S * D);
  _Float16* qPl = alloch((size_t)S * D);
  _Float16* kPh = alloch((size_t)S * D);
  _Float16* kPl = alloch((size_t)S * D);
  _Float16* vPh = alloch((size_t)S * D);
  _Float16* vPl = alloch((size_t)S * D);

  hipMemcpyAsync(h, x, (size_t)S * D * sizeof(float), hipMemcpyDeviceToDevice, stream);

  dim3 blk(256);
  const int ln_grid      = S / 8;
  const int gemm_dd_grid = (S / 16) * (D  / 16) / 8;      // 512
  const int gemm_df_grid = (S / 16) * (FF / 16) / 8;      // 2048
  const int attn_grid    = (S / 16) * H / 8;              // 256
  auto pg = [](int total) { return (total + 255) / 256; };
  const int pa_SD  = pg((S / 16) * (D  / 32) * 32);       // A-pack M=S,K=D
  const int pa_SFF = pg((S / 16) * (FF / 32) * 32);       // A-pack M=S,K=FF
  const int pb_DD  = pg((D  / 16) * (D  / 32) * 32);      // B-pack K=D,N=D
  const int pb_DFF = pg((FF / 16) * (D  / 32) * 32);      // B-pack K=D,N=FF
  const int pb_FFD = pg((D  / 16) * (FF / 32) * 32);      // B-pack K=FF,N=D
  const int pa_qkv = pg((S / 16) * (DH / 32) * 32);       // per-head Q A-pack
  const int pb_kT  = pg((S / 16) * (DH / 32) * 32);       // per-head K^T B-pack
  const int pb_v   = pg((DH / 16) * (S / 32) * 32);       // per-head V B-pack
  const float qscale = 1.0f / sqrtf((float)DH);

  for (int l = 0; l < NLAYER; ++l) {
    const float* lwq = wq + (size_t)l * D * D;  const float* lbq = bq + (size_t)l * D;
    const float* lwk = wk + (size_t)l * D * D;  const float* lbk = bk + (size_t)l * D;
    const float* lwv = wv + (size_t)l * D * D;  const float* lbv = bv + (size_t)l * D;
    const float* lwo = wo + (size_t)l * D * D;  const float* lbo = bo + (size_t)l * D;
    const float* lw1 = w1 + (size_t)l * D * FF; const float* lb1 = b1 + (size_t)l * FF;
    const float* lw2 = w2 + (size_t)l * FF * D; const float* lb2 = b2 + (size_t)l * D;

    // n = LN(h); pack activations once, reused by Q/K/V
    ln_kernel<<<ln_grid, blk, 0, stream>>>(h, nb, S, D);
    pack_a_kernel<<<dim3(pa_SD, 1), blk, 0, stream>>>(nb, aAh, aAl, S, D, D, 1, 0, 1.0f);

    pack_b_kernel<<<dim3(pb_DD, 1), blk, 0, stream>>>(lwq, wBh, wBl, D, D, D, 1, 0);
    gemm_packed_kernel<<<gemm_dd_grid, blk, 0, stream>>>(aAh, aAl, wBh, wBl, lbq, nullptr, qb, S, D, D, 0);
    pack_b_kernel<<<dim3(pb_DD, 1), blk, 0, stream>>>(lwk, wBh, wBl, D, D, D, 1, 0);
    gemm_packed_kernel<<<gemm_dd_grid, blk, 0, stream>>>(aAh, aAl, wBh, wBl, lbk, nullptr, kb, S, D, D, 0);
    pack_b_kernel<<<dim3(pb_DD, 1), blk, 0, stream>>>(lwv, wBh, wBl, D, D, D, 1, 0);
    gemm_packed_kernel<<<gemm_dd_grid, blk, 0, stream>>>(aAh, aAl, wBh, wBl, lbv, nullptr, vb, S, D, D, 0);

    // per-head packs: Q (A-layout, pre-scaled), K^T and V (B-layout)
    pack_a_kernel<<<dim3(pa_qkv, H), blk, 0, stream>>>(qb, qPh, qPl, S, DH, D, 1, DH, qscale);
    pack_b_kernel<<<dim3(pb_kT,  H), blk, 0, stream>>>(kb, kPh, kPl, DH, S, 1, D, DH);
    pack_b_kernel<<<dim3(pb_v,   H), blk, 0, stream>>>(vb, vPh, vPl, S, DH, D, 1, DH);
    attn_kernel<<<attn_grid, blk, 0, stream>>>(qPh, qPl, kPh, kPl, vPh, vPl, ob, S, H, DH, HALFWIN);

    // h = h + o @ wo + bo
    pack_a_kernel<<<dim3(pa_SD, 1), blk, 0, stream>>>(ob, aAh, aAl, S, D, D, 1, 0, 1.0f);
    pack_b_kernel<<<dim3(pb_DD, 1), blk, 0, stream>>>(lwo, wBh, wBl, D, D, D, 1, 0);
    gemm_packed_kernel<<<gemm_dd_grid, blk, 0, stream>>>(aAh, aAl, wBh, wBl, lbo, h, h, S, D, D, 0);

    // h = LN(h); h = h + gelu(h @ w1 + b1) @ w2 + b2  (residual from normalized h)
    ln_kernel<<<ln_grid, blk, 0, stream>>>(h, nb, S, D);
    pack_a_kernel<<<dim3(pa_SD, 1), blk, 0, stream>>>(nb, aAh, aAl, S, D, D, 1, 0, 1.0f);
    pack_b_kernel<<<dim3(pb_DFF, 1), blk, 0, stream>>>(lw1, wBh, wBl, D, FF, FF, 1, 0);
    gemm_packed_kernel<<<gemm_df_grid, blk, 0, stream>>>(aAh, aAl, wBh, wBl, lb1, nullptr, ff, S, FF, D, 1);
    pack_a_kernel<<<dim3(pa_SFF, 1), blk, 0, stream>>>(ff, aAh, aAl, S, FF, FF, 1, 0, 1.0f);
    pack_b_kernel<<<dim3(pb_FFD, 1), blk, 0, stream>>>(lw2, wBh, wBl, FF, D, D, 1, 0);
    gemm_packed_kernel<<<gemm_dd_grid, blk, 0, stream>>>(aAh, aAl, wBh, wBl, lb2, nb, h, S, D, FF, 0);
  }

  hipMemcpyAsync(d_out, h, (size_t)S * D * sizeof(float), hipMemcpyDeviceToDevice, stream);
}
